// Kalei_RNNAgent_1R3_24232205484239
// MI455X (gfx1250) — compile-verified, matrix operand order checked by reading, not run on hardware
//
#include <hip/hip_runtime.h>
#include <hip/hip_bf16.h>
#include <stdint.h>

// ---------------------------------------------------------------------------
// Kalei RNN agent, CDNA5 (gfx1250) implementation.
//   - per-agent masked weights precomputed + packed to bf16 WMMA B-fragments
//   - rows bucketed by agent id so each 16-row tile is agent-uniform
//   - GEMMs via v_wmma_f32_16x16x32_bf16, 4-way register blocking per A-frag
//   - x/h tiles staged with GLOBAL_LOAD_ASYNC_TO_LDS_B128 (ASYNCcnt path)
//   - GRU fully fused in one kernel using 224KB dynamic LDS (320KB/WGP)
// ---------------------------------------------------------------------------

typedef __bf16 bf16_t;
typedef __attribute__((ext_vector_type(16))) __bf16 v16bf;
typedef __attribute__((ext_vector_type(8)))  float  v8f;
typedef __attribute__((ext_vector_type(8)))  unsigned int v8u;

#define HIDDEN 512
#define KDIM   512
#define N_ACT  64

__device__ __forceinline__ v8f wmma_bf16(v16bf a, v16bf b, v8f c) {
  return __builtin_amdgcn_wmma_f32_16x16x32_bf16(
      /*neg_a=*/false, a, /*neg_b=*/false, b,
      /*c_mod=*/(short)0, c, /*reuse_a=*/false, /*reuse_b=*/false);
}

// --- async global -> LDS staging (16KB tile, 128 threads x 8 x 16B) --------
// GLOBAL_LOAD_ASYNC_TO_LDS_B128: VDST = per-lane LDS byte address (low 32 bits
// of the generic shared pointer), VADDR = 64-bit global address; INST_OFFSET
// is added on both sides (ISA 08_async_tensor §4.4).
__device__ __forceinline__ void async_copy_tile16k(bf16_t* lds_dst,
                                                   const bf16_t* gsrc,
                                                   int tid) {
  unsigned lbase = (unsigned)(uintptr_t)(void*)lds_dst + (unsigned)tid * 16u;
  unsigned long long g =
      (unsigned long long)(uintptr_t)gsrc + (unsigned long long)tid * 16ull;
#pragma unroll
  for (int i = 0; i < 8; ++i) {
    asm volatile("global_load_async_to_lds_b128 %0, %1, off offset:%2"
                 :: "v"(lbase), "v"(g), "i"(i * 2048)
                 : "memory");
  }
}
__device__ __forceinline__ void wait_async_lds() {
  asm volatile("s_wait_asynccnt 0" ::: "memory");
}

// A-fragment (16x32 bf16) gather from a row-major [16][512] bf16 LDS tile.
// Layout per ISA 7.12.2: lane L (r=L%16,g=L/16), VGPR j halves (2j,2j+1):
//   k = (j<4?0:16) + 2*(j&3) + (h&1) + 8*g
__device__ __forceinline__ v16bf load_a_frag(const bf16_t* tile, int lane, int kt) {
  const int r = lane & 15;
  const int g = (lane >> 4) & 1;
  const unsigned int* t32 =
      (const unsigned int*)(tile + (size_t)r * KDIM + (size_t)kt * 32);
  v8u u;
#pragma unroll
  for (int j = 0; j < 8; ++j) {
    const int k0 = ((j < 4) ? 0 : 16) + 2 * (j & 3) + 8 * g;  // even
    u[j] = t32[k0 >> 1];
  }
  return __builtin_bit_cast(v16bf, u);
}

__device__ __forceinline__ v16bf load_b_frag(const bf16_t* Bp, int lane,
                                             int tileOff, int kt) {
  return *(const v16bf*)(Bp + ((size_t)tileOff * 16 + kt) * 512 + lane * 16);
}

// Packed B index for element (a, o, i) of an [A][O][512] weight tensor.
// Mirror of the A layout with lanes carrying the N (output) dimension.
__device__ __forceinline__ size_t pack_index(int a, int o, int i, int O) {
  const int ct = o >> 4, n = o & 15;
  const int kt = i >> 5, k = i & 31;
  const int g  = (k >> 3) & 1;
  const int hi = (k >> 4) & 1;
  const int j  = hi * 4 + ((k & 7) >> 1);
  const int h  = 2 * j + (k & 1);
  const int lane = g * 16 + n;
  return ((((size_t)a * (O >> 4) + ct) * (KDIM >> 5) + kt) * 32 + lane) * 16 + h;
}

// ---------------------------------------------------------------------------
// Weight mask + pack: Wm = sign(W) * relu(|W| - sigmoid(t_a)), bf16, swizzled.
// T == nullptr -> plain convert (GRU weights).
// ---------------------------------------------------------------------------
__global__ void pack_weights_kernel(const float* __restrict W,
                                    const float* __restrict T,
                                    bf16_t* __restrict out,
                                    int A, int O) {
  const size_t total = (size_t)A * O * KDIM;
  for (size_t idx = (size_t)blockIdx.x * blockDim.x + threadIdx.x; idx < total;
       idx += (size_t)gridDim.x * blockDim.x) {
    const int i = (int)(idx % KDIM);
    const size_t rem = idx / KDIM;
    const int o = (int)(rem % O);
    const int a = (int)(rem / O);
    const float w = W[(size_t)o * KDIM + i];
    float v;
    if (T) {
      const float t = T[idx];                      // [A,O,K] contiguous
      const float s = 1.0f / (1.0f + __expf(-t));  // sigmoid(t)
      const float m = fabsf(w) - s;
      v = (m > 0.0f) ? copysignf(m, w) : 0.0f;
    } else {
      v = w;
    }
    out[pack_index(a, o, i, O)] = (bf16_t)v;
  }
}

// ---------------------------------------------------------------------------
// Agent bucketing (counting sort, buckets padded to multiples of 16 rows)
// ---------------------------------------------------------------------------
__global__ void count_kernel(const int* __restrict ids, int* __restrict counts,
                             int N) {
  const int n = blockIdx.x * blockDim.x + threadIdx.x;
  if (n < N) atomicAdd(&counts[ids[n] & 7], 1);
}

__global__ void offsets_kernel(const int* __restrict counts,
                               int* __restrict offsets,
                               int* __restrict tile_agent, int ntiles) {
  if (blockIdx.x == 0 && threadIdx.x == 0) {
    int ends[8];
    int cum = 0;
    for (int a = 0; a < 8; ++a) {
      offsets[a] = cum;
      cum += (counts[a] + 15) & ~15;
      ends[a] = cum;
    }
    for (int t = 0; t < ntiles; ++t) {
      const int row = t * 16;
      int ag = 7;
      for (int a = 0; a < 8; ++a)
        if (row < ends[a]) { ag = a; break; }
      tile_agent[t] = ag;
    }
  }
}

__global__ void assign_kernel(const int* __restrict ids,
                              const int* __restrict offsets,
                              int* __restrict cursor,
                              int* __restrict row_index, int N) {
  const int n = blockIdx.x * blockDim.x + threadIdx.x;
  if (n < N) {
    const int a = ids[n] & 7;
    const int pos = offsets[a] + atomicAdd(&cursor[a], 1);
    row_index[pos] = n;
  }
}

__global__ void gather_kernel(const float* __restrict x,
                              const float* __restrict h,
                              const int* __restrict row_index,
                              bf16_t* __restrict xs, bf16_t* __restrict hs,
                              int npad) {
  const size_t idx = (size_t)blockIdx.x * blockDim.x + threadIdx.x;
  const size_t total = (size_t)npad * KDIM;
  if (idx < total) {
    const int r = (int)(idx >> 9);
    const int c = (int)(idx & 511);
    const int orig = row_index[r];
    if (orig >= 0) {
      xs[idx] = (bf16_t)x[(size_t)orig * KDIM + c];
      hs[idx] = (bf16_t)h[(size_t)orig * KDIM + c];
    } else {
      xs[idx] = (bf16_t)0.0f;
      hs[idx] = (bf16_t)0.0f;
    }
  }
}

// ---------------------------------------------------------------------------
// Kalei FC layer (512->512): y = relu(x @ Wm[agent]^T + b), bf16 out.
// One block per 16-row tile, 4 waves x 8 column tiles in 2 groups of 4
// (4 accumulators share each A fragment -> 4x fewer DS reads).
// ---------------------------------------------------------------------------
__device__ __forceinline__ void store_relu_bf16(v8f acc, int tile, int ct,
                                                int lane,
                                                const float* __restrict bias,
                                                bf16_t* __restrict out) {
  const int col = ct * 16 + (lane & 15);
  const float bv = bias[col];
  const int rbase = (lane < 16) ? 0 : 8;
#pragma unroll
  for (int d = 0; d < 8; ++d) {
    const float v = fmaxf(acc[d] + bv, 0.0f);
    out[((size_t)tile * 16 + rbase + d) * HIDDEN + col] = (bf16_t)v;
  }
}

__global__ void __launch_bounds__(128)
kalei_layer_kernel(const bf16_t* __restrict xs, const bf16_t* __restrict Wp,
                   const float* __restrict bias,
                   const int* __restrict tile_agent,
                   bf16_t* __restrict out) {
  __shared__ bf16_t xt[16 * KDIM];  // 16 KB
  const int tile = blockIdx.x;
  const int tid = threadIdx.x;

  async_copy_tile16k(xt, xs + (size_t)tile * 16 * KDIM, tid);
  wait_async_lds();
  __syncthreads();

  const int agent = tile_agent[tile];
  const int wave = tid >> 5;
  const int lane = tid & 31;

#pragma unroll
  for (int grp = 0; grp < 2; ++grp) {
    const int ct0 = wave * 8 + grp * 4;
    const bf16_t* Bp = Wp + (((size_t)agent * 32 + ct0) * 16) * 512;
    v8f a0 = {0.f, 0.f, 0.f, 0.f, 0.f, 0.f, 0.f, 0.f};
    v8f a1 = a0, a2 = a0, a3 = a0;
#pragma unroll 2
    for (int kt = 0; kt < 16; ++kt) {
      const v16bf a = load_a_frag(xt, lane, kt);
      const v16bf b0 = load_b_frag(Bp, lane, 0, kt);
      const v16bf b1 = load_b_frag(Bp, lane, 1, kt);
      const v16bf b2 = load_b_frag(Bp, lane, 2, kt);
      const v16bf b3 = load_b_frag(Bp, lane, 3, kt);
      a0 = wmma_bf16(a, b0, a0);
      a1 = wmma_bf16(a, b1, a1);
      a2 = wmma_bf16(a, b2, a2);
      a3 = wmma_bf16(a, b3, a3);
    }
    store_relu_bf16(a0, tile, ct0 + 0, lane, bias, out);
    store_relu_bf16(a1, tile, ct0 + 1, lane, bias, out);
    store_relu_bf16(a2, tile, ct0 + 2, lane, bias, out);
    store_relu_bf16(a3, tile, ct0 + 3, lane, bias, out);
  }
}

// ---------------------------------------------------------------------------
// Fused GRU: gi = x1 @ Wih^T + bih, gh = h @ Whh^T + bhh (both 16x1536 in LDS,
// f32), then elementwise gate math. 224KB dynamic LDS (CDNA5: 320KB/WGP).
// Waves 0,1 -> gi column tiles 0..95; waves 2,3 -> gh column tiles 0..95.
// ---------------------------------------------------------------------------
__global__ void __launch_bounds__(128)
gru_fused_kernel(const bf16_t* __restrict x1s, const bf16_t* __restrict hs,
                 const bf16_t* __restrict Wih, const bf16_t* __restrict Whh,
                 const float* __restrict bih, const float* __restrict bhh,
                 const float* __restrict hidden_in,   // original f32 [N,512]
                 const int* __restrict row_index,
                 float* __restrict hout,              // f32 [N,512] (d_out)
                 bf16_t* __restrict hnew_sorted) {
  extern __shared__ char smem[];
  bf16_t* xt = (bf16_t*)smem;                        // 16KB
  bf16_t* ht = (bf16_t*)(smem + 16384);              // 16KB
  float* gi = (float*)(smem + 32768);                // 96KB
  float* gh = (float*)(smem + 32768 + 98304);        // 96KB

  const int tile = blockIdx.x;
  const int tid = threadIdx.x;

  async_copy_tile16k(xt, x1s + (size_t)tile * 16 * KDIM, tid);
  async_copy_tile16k(ht, hs + (size_t)tile * 16 * KDIM, tid);
  wait_async_lds();
  __syncthreads();

  const int wave = tid >> 5;
  const int lane = tid & 31;

  const bool isGi = (wave < 2);
  const bf16_t* tileA = isGi ? xt : ht;
  const bf16_t* Wp = isGi ? Wih : Whh;
  const float* bb = isGi ? bih : bhh;
  float* gout = isGi ? gi : gh;
  const int ctbase = (wave & 1) * 48;  // each wave: 48 column tiles of 16

#pragma unroll 1
  for (int g = 0; g < 12; ++g) {
    const int ct0 = ctbase + g * 4;
    const bf16_t* Bp = Wp + ((size_t)ct0 * 16) * 512;
    v8f a0 = {0.f, 0.f, 0.f, 0.f, 0.f, 0.f, 0.f, 0.f};
    v8f a1 = a0, a2 = a0, a3 = a0;
#pragma unroll 2
    for (int kt = 0; kt < 16; ++kt) {
      const v16bf a = load_a_frag(tileA, lane, kt);
      const v16bf b0 = load_b_frag(Bp, lane, 0, kt);
      const v16bf b1 = load_b_frag(Bp, lane, 1, kt);
      const v16bf b2 = load_b_frag(Bp, lane, 2, kt);
      const v16bf b3 = load_b_frag(Bp, lane, 3, kt);
      a0 = wmma_bf16(a, b0, a0);
      a1 = wmma_bf16(a, b1, a1);
      a2 = wmma_bf16(a, b2, a2);
      a3 = wmma_bf16(a, b3, a3);
    }
    const int rbase = (lane < 16) ? 0 : 8;
    v8f accs[4] = {a0, a1, a2, a3};
#pragma unroll
    for (int q = 0; q < 4; ++q) {
      const int col = (ct0 + q) * 16 + (lane & 15);
      const float bv = bb[col];
#pragma unroll
      for (int d = 0; d < 8; ++d)
        gout[(rbase + d) * 1536 + col] = accs[q][d] + bv;
    }
  }
  __syncthreads();

  // elementwise GRU combine: r/z/n gates, new hidden
  for (int idx = tid; idx < 16 * HIDDEN; idx += 128) {
    const int r = idx >> 9;
    const int c = idx & 511;
    const int srow = tile * 16 + r;
    const float ir = gi[r * 1536 + c];
    const float iz = gi[r * 1536 + 512 + c];
    const float in_ = gi[r * 1536 + 1024 + c];
    const float hr = gh[r * 1536 + c];
    const float hz = gh[r * 1536 + 512 + c];
    const float hn = gh[r * 1536 + 1024 + c];
    const float rg = 1.0f / (1.0f + __expf(-(ir + hr)));
    const float zg = 1.0f / (1.0f + __expf(-(iz + hz)));
    const float ng = tanhf(in_ + rg * hn);
    const int orig = row_index[srow];
    const float hp = (orig >= 0) ? hidden_in[(size_t)orig * HIDDEN + c] : 0.0f;
    const float hv = (1.0f - zg) * ng + zg * hp;
    hnew_sorted[(size_t)srow * HIDDEN + c] = (bf16_t)hv;
    if (orig >= 0) hout[(size_t)orig * HIDDEN + c] = hv;
  }
}

// ---------------------------------------------------------------------------
// Kalei FC4 (512->64): q = x @ Wm4[agent]^T + b, f32 scattered to d_out.
// 4 waves = 4 column tiles of 16.
// ---------------------------------------------------------------------------
__global__ void __launch_bounds__(128)
fc4_kernel(const bf16_t* __restrict xs, const bf16_t* __restrict Wp,
           const float* __restrict bias, const int* __restrict tile_agent,
           const int* __restrict row_index, float* __restrict qout) {
  __shared__ bf16_t xt[16 * KDIM];
  const int tile = blockIdx.x;
  const int tid = threadIdx.x;

  async_copy_tile16k(xt, xs + (size_t)tile * 16 * KDIM, tid);
  wait_async_lds();
  __syncthreads();

  const int agent = tile_agent[tile];
  const int wave = tid >> 5;
  const int lane = tid & 31;
  const int ct = wave;
  const bf16_t* Bp = Wp + (((size_t)agent * 4 + ct) * 16) * 512;
  v8f acc = {0.f, 0.f, 0.f, 0.f, 0.f, 0.f, 0.f, 0.f};
#pragma unroll 4
  for (int kt = 0; kt < 16; ++kt) {
    const v16bf a = load_a_frag(xt, lane, kt);
    const v16bf b = load_b_frag(Bp, lane, 0, kt);
    acc = wmma_bf16(a, b, acc);
  }
  const int col = ct * 16 + (lane & 15);
  const float bv = bias[col];
  const int rbase = (lane < 16) ? 0 : 8;
#pragma unroll
  for (int d = 0; d < 8; ++d) {
    const int srow = tile * 16 + rbase + d;
    const int orig = row_index[srow];
    if (orig >= 0) qout[(size_t)orig * N_ACT + col] = acc[d] + bv;
  }
}

// ---------------------------------------------------------------------------
// Host side
// ---------------------------------------------------------------------------
static inline size_t align256(size_t x) { return (x + 255) & ~(size_t)255; }

extern "C" void kernel_launch(void* const* d_in, const int* in_sizes, int n_in,
                              void* d_out, int out_size, void* d_ws,
                              size_t ws_size, hipStream_t stream) {
  const float* inputs   = (const float*)d_in[0];
  const float* hidden   = (const float*)d_in[1];
  const int*   ids      = (const int*)d_in[2];
  const float* fc1_w = (const float*)d_in[3];
  const float* fc1_b = (const float*)d_in[4];
  const float* fc1_t = (const float*)d_in[5];
  const float* gru_wih = (const float*)d_in[6];
  const float* gru_whh = (const float*)d_in[7];
  const float* gru_bih = (const float*)d_in[8];
  const float* gru_bhh = (const float*)d_in[9];
  const float* fc2_w = (const float*)d_in[10];
  const float* fc2_b = (const float*)d_in[11];
  const float* fc2_t = (const float*)d_in[12];
  const float* fc3_w = (const float*)d_in[13];
  const float* fc3_b = (const float*)d_in[14];
  const float* fc3_t = (const float*)d_in[15];
  const float* fc4_w = (const float*)d_in[16];
  const float* fc4_b = (const float*)d_in[17];
  const float* fc4_t = (const float*)d_in[18];

  const int N = in_sizes[2];            // B*A = 16384 rows
  const int NT = N / 16 + 8;            // worst-case tiles after 16-padding
  const int NPAD = NT * 16;

  float* q_out = (float*)d_out;                 // [N, 64]
  float* h_out = q_out + (size_t)N * N_ACT;     // [N, 512]

  // workspace carve-up
  char* ws = (char*)d_ws;
  size_t off = 0;
  auto take = [&](size_t bytes) { char* p = ws + off; off += align256(bytes); return p; };
  bf16_t* Wp1 = (bf16_t*)take((size_t)8 * 512 * 512 * 2);
  bf16_t* Wp2 = (bf16_t*)take((size_t)8 * 512 * 512 * 2);
  bf16_t* Wp3 = (bf16_t*)take((size_t)8 * 512 * 512 * 2);
  bf16_t* Wp4 = (bf16_t*)take((size_t)8 * 64 * 512 * 2);
  bf16_t* Wih = (bf16_t*)take((size_t)1536 * 512 * 2);
  bf16_t* Whh = (bf16_t*)take((size_t)1536 * 512 * 2);
  int* counts   = (int*)take(8 * 4);
  int* offsets  = (int*)take(8 * 4);
  int* cursor   = (int*)take(8 * 4);
  int* tile_agent = (int*)take((size_t)NT * 4);
  int* row_index  = (int*)take((size_t)NPAD * 4);
  bf16_t* x_sorted = (bf16_t*)take((size_t)NPAD * 512 * 2);  // reused as act2
  bf16_t* h_sorted = (bf16_t*)take((size_t)NPAD * 512 * 2);  // reused as h'
  bf16_t* act1     = (bf16_t*)take((size_t)NPAD * 512 * 2);  // reused as act3
  bf16_t* act2 = x_sorted;
  bf16_t* act3 = act1;
  (void)ws_size;

  hipMemsetAsync(counts, 0, 8 * 4, stream);
  hipMemsetAsync(cursor, 0, 8 * 4, stream);
  hipMemsetAsync(row_index, 0xFF, (size_t)NPAD * 4, stream);  // -1

  // weight mask + pack (redone every call; deterministic)
  pack_weights_kernel<<<2048, 256, 0, stream>>>(fc1_w, fc1_t, Wp1, 8, 512);
  pack_weights_kernel<<<2048, 256, 0, stream>>>(fc2_w, fc2_t, Wp2, 8, 512);
  pack_weights_kernel<<<2048, 256, 0, stream>>>(fc3_w, fc3_t, Wp3, 8, 512);
  pack_weights_kernel<<<1024, 256, 0, stream>>>(fc4_w, fc4_t, Wp4, 8, 64);
  pack_weights_kernel<<<1024, 256, 0, stream>>>(gru_wih, nullptr, Wih, 1, 1536);
  pack_weights_kernel<<<1024, 256, 0, stream>>>(gru_whh, nullptr, Whh, 1, 1536);

  // bucket rows by agent
  count_kernel<<<(N + 255) / 256, 256, 0, stream>>>(ids, counts, N);
  offsets_kernel<<<1, 32, 0, stream>>>(counts, offsets, tile_agent, NT);
  assign_kernel<<<(N + 255) / 256, 256, 0, stream>>>(ids, offsets, cursor, row_index, N);
  {
    const size_t total = (size_t)NPAD * 512;
    gather_kernel<<<(unsigned)((total + 255) / 256), 256, 0, stream>>>(
        inputs, hidden, row_index, x_sorted, h_sorted, NPAD);
  }

  // fc1: relu(kalei(x))
  kalei_layer_kernel<<<NT, 128, 0, stream>>>(x_sorted, Wp1, fc1_b, tile_agent, act1);

  // fused GRU (224KB dynamic LDS per workgroup)
  const int GRU_SMEM = 16384 * 2 + 98304 * 2;  // 229376 B
  hipFuncSetAttribute((const void*)gru_fused_kernel,
                      hipFuncAttributeMaxDynamicSharedMemorySize, GRU_SMEM);
  gru_fused_kernel<<<NT, 128, GRU_SMEM, stream>>>(
      act1, h_sorted, Wih, Whh, gru_bih, gru_bhh, hidden, row_index,
      h_out, /*hnew in-place*/ h_sorted);

  // fc2, fc3: relu(kalei(h)), relu(kalei(q))
  kalei_layer_kernel<<<NT, 128, 0, stream>>>(h_sorted, Wp2, fc2_b, tile_agent, act2);
  kalei_layer_kernel<<<NT, 128, 0, stream>>>(act2, Wp3, fc3_b, tile_agent, act3);

  // fc4: q (no activation), scatter f32 to d_out
  fc4_kernel<<<NT, 128, 0, stream>>>(act3, Wp4, fc4_b, tile_agent, row_index, q_out);
  (void)n_in; (void)out_size;
}
